// Transformer_CS_86629490360502
// MI455X (gfx1250) — matrix operand order repaired and checked by reading, not compiled, not run on hardware
//
#include <hip/hip_runtime.h>
#include <hip/hip_bf16.h>

typedef __attribute__((ext_vector_type(16))) _Float16 v16h;
typedef __attribute__((ext_vector_type(8)))  _Float16 v8h;
typedef __attribute__((ext_vector_type(2)))  _Float16 v2h;
typedef __attribute__((ext_vector_type(8)))  float    v8f;
typedef __attribute__((ext_vector_type(4)))  float    f32x4;
typedef __attribute__((ext_vector_type(4)))  unsigned int u32x4;
typedef __attribute__((ext_vector_type(4)))  int      i32x4;
typedef __attribute__((ext_vector_type(8)))  int      i32x8;

#define L_ALL 704
#define NB    64
#define NROW  (L_ALL * NB)   // 45056
#define DM    256
#define DFF   2048
#define SCALE 0.17677669529663687f   // 32^-0.5

#if defined(__has_builtin)
#  if __has_builtin(__builtin_amdgcn_tensor_load_to_lds)
#    define HAVE_TDM 1
#  else
#    define HAVE_TDM 0
#  endif
#else
#  define HAVE_TDM 0
#endif

// ---------------------------------------------------------------------------
// CDNA5 16-bit WMMA fragments as used here are K-contiguous per lane (not the ISA table order):
//   A (16x32): lane&15 = M; halves [kb0, kb0+16), kb0 = (lane<16)?0:16
//   B (32x16): lane&15 = N; halves [kb0, kb0+16), kb0 = (lane<16)?0:16
//   C/D f32  : VGPR r -> M = r + (lane<16?0:8), N = lane&15
// So fragments are assembled from 2 x b128 loads when source rows are K-major.
// ---------------------------------------------------------------------------
__device__ __forceinline__ v16h frag_cat(v8h lo, v8h hi) {
    return __builtin_shufflevector(lo, hi, 0, 1, 2, 3, 4, 5, 6, 7,
                                           8, 9, 10, 11, 12, 13, 14, 15);
}

// ---------------------------------------------------------------------------
// Tensor Data Mover: 2D tile (tile_w dwords x tile_h rows) global -> LDS.
// D# per ISA ch.8: group0 {count=1, lds_addr, global_addr, type=2},
// group1 {data_size=4B, pad cfg, dims, tile dims, row stride}.
// ---------------------------------------------------------------------------
__device__ __forceinline__ void tdm_load_2d(unsigned int lds_addr, const void* gptr,
                                            unsigned int tile_w_du, unsigned int tile_h,
                                            unsigned long long row_stride_du,
                                            unsigned int pad_interval_code,
                                            unsigned int pad_amount_code,
                                            bool pad_en) {
#if HAVE_TDM
    unsigned long long ga = (unsigned long long)(uintptr_t)gptr;
    u32x4 g0;
    g0[0] = 1u;                                          // count=1
    g0[1] = lds_addr;                                    // lds_addr
    g0[2] = (unsigned int)(ga & 0xFFFFFFFFu);            // global_addr[31:0]
    g0[3] = (unsigned int)((ga >> 32) & 0x1FFFFFFu) | (2u << 30);  // [56:32] + type=2
    unsigned int w0 = (2u << 16);                        // data_size = 4B
    if (pad_en) w0 |= (1u << 20) | (pad_interval_code << 22) | (pad_amount_code << 25);
    unsigned int td0 = (unsigned int)row_stride_du;      // tensor_dim0 (OOB bound)
    unsigned int td1 = tile_h;                           // tensor_dim1
    i32x8 g1;
    g1[0] = (int)w0;
    g1[1] = (int)((td0 & 0xFFFFu) << 16);                // td0[15:0] -> bits 63:48
    g1[2] = (int)(((td0 >> 16) & 0xFFFFu) | ((td1 & 0xFFFFu) << 16));
    g1[3] = (int)(((td1 >> 16) & 0xFFFFu) | ((tile_w_du & 0xFFFFu) << 16)); // tile_dim0
    g1[4] = (int)(tile_h & 0xFFFFu);                     // tile_dim1 (tile_dim2 = 0)
    g1[5] = (int)(unsigned int)(row_stride_du & 0xFFFFFFFFu);      // dim0 stride lo
    g1[6] = (int)(unsigned int)((row_stride_du >> 32) & 0xFFFFu);  // stride hi
    g1[7] = 0;
    i32x4 z = {0, 0, 0, 0};
#if __clang_major__ >= 23
    i32x8 z8 = {0, 0, 0, 0, 0, 0, 0, 0};
    __builtin_amdgcn_tensor_load_to_lds(g0, g1, z, z, z8, 0);
#else
    __builtin_amdgcn_tensor_load_to_lds(g0, g1, z, z, 0);
#endif
#endif
}

__device__ __forceinline__ void tdm_wait() {
#if HAVE_TDM
    __builtin_amdgcn_s_wait_tensorcnt(0);
#endif
}

// ---------------------------------------------------------------------------
// f32 -> f16 conversion, vectorized (b128 in / b128 out)
// ---------------------------------------------------------------------------
__global__ void k_cvt_f16(const float* __restrict__ in, _Float16* __restrict__ out, int n) {
    int n8 = n >> 3;
    for (int i = blockIdx.x * blockDim.x + threadIdx.x; i < n8; i += gridDim.x * blockDim.x) {
        const f32x4* p = (const f32x4*)(in + (size_t)i * 8);
        f32x4 a = p[0], b = p[1];
        v8h o;
#pragma unroll
        for (int u = 0; u < 4; ++u) { o[u] = (_Float16)a[u]; o[4 + u] = (_Float16)b[u]; }
        *(v8h*)(out + (size_t)i * 8) = o;
    }
}

// ---------------------------------------------------------------------------
// Window permutation tables + shuffle-retention mask (one block per head).
// ---------------------------------------------------------------------------
__device__ __forceinline__ int window_token(int wi, int p, int q, int ws, int shift) {
    int n8 = (8 / ws) * (8 / ws);
    if (wi < n8) {
        int g = 8 / ws;
        return ((wi / g) * ws + p) * 8 + (wi % g) * ws + q;
    } else if (wi < 2 * n8) {
        int g = 8 / ws; wi -= n8;
        return 64 + ((wi / g) * ws + p) * 8 + (wi % g) * ws + q;
    } else {
        int g = 24 / ws; wi -= 2 * n8;
        int row = ((wi / g) * ws + p + shift) % 24;
        int col = ((wi % g) * ws + q + shift) % 24;
        return 128 + row * 24 + col;
    }
}

__global__ void k_build_tables(int* __restrict__ qtok, int* __restrict__ kstok,
                               float* __restrict__ srm) {
    const int wsArr[8] = {1, 2, 4, 8, 1, 2, 4, 8};
    int h = blockIdx.x;
    int ws = wsArr[h];
    int shift = (h >= 4) ? (ws / 2) : 0;
    int ws2 = ws * ws;
    for (int j = threadIdx.x; j < L_ALL; j += blockDim.x) {
        int wi = j / ws2, pq = j % ws2;
        qtok[h * L_ALL + j] = window_token(wi, pq / ws, pq % ws, ws, shift);
    }
    for (int jw = threadIdx.x; jw < L_ALL; jw += blockDim.x) {
        int j = jw / ws2, s = jw % ws2;
        int u1 = s / ws, v1 = s % ws;
        float wu = (float)u1 / (float)ws; wu = (wu > 0.5f) ? wu : 1.0f - wu;
        float wv = (float)v1 / (float)ws; wv = (wv > 0.5f) ? wv : 1.0f - wv;
        srm[h * L_ALL + jw] = (ws == 1) ? 0.0f : (wu * wv - 1.0f);
        for (int t = 0; t < ws2; ++t) {
            int pr = (t / ws - u1 + ws) % ws, qc = (t % ws - v1 + ws) % ws;
            kstok[(h * L_ALL + jw) * 64 + t] = window_token(j, pr, qc, ws, shift);
        }
    }
}

// ---------------------------------------------------------------------------
// Tiled f16 WMMA GEMM. WG tile 64x128, 8 waves (wave tile 16x64), K step 32.
// A tile: TDM async global->LDS, double-buffered, padded rows (20 dwords).
// B tile: cooperative transposed staging [N][K] (packed b32 stores, bank-clean).
// All 10 fragment b128 LDS loads are issued before the 4-WMMA chain so the
// scheduler can overlap LDS latency with matrix issue (partial dscnt waits).
// ---------------------------------------------------------------------------
__global__ void k_gemm_f16(const _Float16* __restrict__ A, const _Float16* __restrict__ B,
                           const float* __restrict__ bias, int M, int N, int K, int mode,
                           float* __restrict__ outF, _Float16* __restrict__ outH,
                           const float* __restrict__ resid,
                           _Float16* __restrict__ q16, _Float16* __restrict__ k16,
                           _Float16* __restrict__ v16) {
    __shared__ _Float16 As[2][64 * 40];   // rows of 32 halves + 8 pad (TDM pad 16B/64B)
    __shared__ _Float16 Bs[128][40];      // transposed [n][k], stride 40 halves

    const int tid  = threadIdx.x;
    const int lane = tid & 31;
    const int wave = tid >> 5;
    const int wm   = wave & 3;
    const int wn   = wave >> 2;
    const int m0   = blockIdx.y * 64;
    const int n0   = blockIdx.x * 128;
    const int rp   = tid & 15;            // B staging: row pair
    const int cg   = tid >> 4;            // B staging: col group of 8

    v8f acc[4];
#pragma unroll
    for (int j = 0; j < 4; ++j)
#pragma unroll
        for (int r = 0; r < 8; ++r) acc[j][r] = 0.0f;

    if (wave == 0) {
        tdm_load_2d((unsigned int)(uintptr_t)&As[0][0], &A[(size_t)m0 * K],
                    16, 64, (unsigned long long)(K >> 1), 3, 3, true);
        tdm_wait();
    }
#if !HAVE_TDM
    {   // fallback A staging for k=0
        int row = tid >> 2, ch = tid & 3;
        *(v8h*)&As[0][row * 40 + ch * 8] = *(const v8h*)&A[(size_t)(m0 + row) * K + ch * 8];
    }
#endif

    int cur = 0;
    for (int kk = 0; kk < K; kk += 32) {
        // ---- stage B tile transposed: Bs[n][k] ----
        {
            const _Float16* g = &B[(size_t)(kk + 2 * rp) * N + n0 + cg * 8];
            v8h b0 = *(const v8h*)g;
            v8h b1 = *(const v8h*)(g + N);
#pragma unroll
            for (int i = 0; i < 8; ++i) {
                v2h pr2 = {b0[i], b1[i]};
                *(v2h*)&Bs[cg * 8 + i][2 * rp] = pr2;
            }
        }
        // ---- async A for next K step ----
        if (kk + 32 < K) {
            if (wave == 0)
                tdm_load_2d((unsigned int)(uintptr_t)&As[cur ^ 1][0],
                            &A[(size_t)m0 * K + kk + 32],
                            16, 64, (unsigned long long)(K >> 1), 3, 3, true);
#if !HAVE_TDM
            int row = tid >> 2, ch = tid & 3;
            *(v8h*)&As[cur ^ 1][row * 40 + ch * 8] =
                *(const v8h*)&A[(size_t)(m0 + row) * K + kk + 32 + ch * 8];
#endif
            __builtin_prefetch(&B[(size_t)(kk + 32 + 2 * rp) * N + n0 + cg * 8], 0, 1);
        }
        __syncthreads();

        // ---- all fragment loads first, then back-to-back WMMAs ----
        const int m   = lane & 15;
        const int khi = (lane < 16) ? 0 : 8;
        const int kb0 = (lane < 16) ? 0 : 16;
        const _Float16* ar = &As[cur][(wm * 16 + m) * 40];
        v16h af = frag_cat(*(const v8h*)(ar + kb0), *(const v8h*)(ar + kb0 + 8));

        v16h bf[4];
#pragma unroll
        for (int j = 0; j < 4; ++j) {
            const _Float16* br = &Bs[wn * 64 + j * 16 + (lane & 15)][kb0];
            bf[j] = frag_cat(*(const v8h*)br, *(const v8h*)(br + 8));
        }
#pragma unroll
        for (int j = 0; j < 4; ++j)
            acc[j] = __builtin_amdgcn_wmma_f32_16x16x32_f16(
                false, af, false, bf[j], (short)0, acc[j], false, false);

        if (wave == 0) tdm_wait();
        __syncthreads();
        cur ^= 1;
    }

    // ---- epilogue ----
    const int n    = lane & 15;
    const int moff = (lane < 16) ? 0 : 8;
#pragma unroll
    for (int j = 0; j < 4; ++j) {
#pragma unroll
        for (int r = 0; r < 8; ++r) {
            int row = m0 + wm * 16 + r + moff;
            int col = n0 + wn * 64 + j * 16 + n;
            float val = acc[j][r] + bias[col];
            if (mode == 0) {
                int part = col >> 8;
                int hh = (col >> 5) & 7, c = col & 31;
                int l = row >> 6, bb = row & 63;
                size_t idx = ((size_t)(bb * 8 + hh) * L_ALL + l) * 32 + c;
                if (part == 0)      q16[idx] = (_Float16)(val * SCALE);
                else if (part == 1) k16[idx] = (_Float16)val;
                else                v16[idx] = (_Float16)val;
            } else if (mode == 2) {
                outH[(size_t)row * N + col] = (_Float16)fmaxf(val, 0.0f);
            } else {
                outF[(size_t)row * N + col] = val + resid[(size_t)row * N + col];
            }
        }
    }
}

// ---------------------------------------------------------------------------
// Per-head windowed flash attention. WG = (batch, head, 16-query-window tile),
// 512 threads = 16 waves. Q/K fragments = direct v8h global loads (rows are
// 64B-aligned K-major); P fragments = b128 LDS float4 loads + pack-convert.
// ---------------------------------------------------------------------------
__global__ void k_attention(const _Float16* __restrict__ q16, const _Float16* __restrict__ k16,
                            const _Float16* __restrict__ v16, _Float16* __restrict__ attn16,
                            const int* __restrict__ qtok, const int* __restrict__ kstok,
                            const float* __restrict__ srm, const float* __restrict__ rel,
                            int head, int ws, int L) {
    __shared__ float S[16][716];   // stride 716: 16B-aligned rows, conflict-free b128

    const int ws2 = ws * ws;
    const int qTiles = (L + 15) / 16;
    const int b  = blockIdx.x / qTiles;
    const int qt = blockIdx.x % qTiles;
    const int lane = threadIdx.x & 31;
    const int wave = threadIdx.x >> 5;

    const _Float16* qh = q16 + ((size_t)(b * 8 + head) * L_ALL) * 32;
    const _Float16* kh = k16 + ((size_t)(b * 8 + head) * L_ALL) * 32;
    const _Float16* vh = v16 + ((size_t)(b * 8 + head) * L_ALL) * 32;
    const int*   qtokH  = qtok  + head * L_ALL;
    const int*   kstokH = kstok + head * L_ALL * 64;
    const float* srmH   = srm   + head * L_ALL;

    const int khi = (lane < 16) ? 0 : 8;
    const int kb0 = (lane < 16) ? 0 : 16;

    // ---- Stage 1: S = Qp @ Ks^T (44 key blocks over 16 waves) ----
    for (int jb = wave; jb < 44; jb += 16) {
        v8f acc;
#pragma unroll
        for (int r = 0; r < 8; ++r) acc[r] = 0.0f;

        for (int kc = 0; kc < ws2; ++kc) {
            int m = lane & 15;
            int row = qt * 16 + m;
            bool valid = row < L;
            const _Float16* qp = qh + (size_t)(valid ? qtokH[row * ws2 + kc] : 0) * 32;
            int jw = jb * 16 + (lane & 15);
            const _Float16* kp = kh + (size_t)kstokH[jw * 64 + kc] * 32;
            // issue all 4 global b128 loads before packing
            v8h qa = *(const v8h*)(qp + kb0);
            v8h qb = *(const v8h*)(qp + kb0 + 8);
            v8h ka = *(const v8h*)(kp + kb0);
            v8h kb = *(const v8h*)(kp + kb0 + 8);
            v16h af = frag_cat(qa, qb);
            if (!valid) {
#pragma unroll
                for (int i = 0; i < 16; ++i) af[i] = (_Float16)0.0f;
            }
            v16h bfr = frag_cat(ka, kb);
            acc = __builtin_amdgcn_wmma_f32_16x16x32_f16(
                false, af, false, bfr, (short)0, acc, false, false);
        }
        int n = lane & 15;
        int moff = (lane < 16) ? 0 : 8;
#pragma unroll
        for (int r = 0; r < 8; ++r) S[r + moff][jb * 16 + n] = acc[r];
    }
    __syncthreads();

    // ---- Softmax: wave per row; score = S/ws^2 + srm + rel ----
    {
        int row = wave;
        int i = qt * 16 + row;
        if (i < L) {
            const float inv = 1.0f / (float)ws2;
            float mx = -1e30f;
            for (int col = lane; col < L_ALL; col += 32) {
                float s = S[row][col] * inv + srmH[col] + rel[(size_t)i * L_ALL + col];
                S[row][col] = s;
                mx = fmaxf(mx, s);
            }
#pragma unroll
            for (int o = 16; o > 0; o >>= 1) mx = fmaxf(mx, __shfl_xor(mx, o, 32));
            float sum = 0.0f;
            for (int col = lane; col < L_ALL; col += 32) {
                float p = __expf(S[row][col] - mx);
                S[row][col] = p;
                sum += p;
            }
#pragma unroll
            for (int o = 16; o > 0; o >>= 1) sum += __shfl_xor(sum, o, 32);
            float rs = 1.0f / sum;
            for (int col = lane; col < L_ALL; col += 32) S[row][col] *= rs;
        } else {
            for (int col = lane; col < L_ALL; col += 32) S[row][col] = 0.0f;
        }
    }
    __syncthreads();

    // ---- Stage 2: out = P @ Vs ----
    const int ntiles = 2 * ws2;
    for (int nt = wave; nt < ntiles; nt += 16) {
        int nc0 = nt * 16;
        int tn = nc0 >> 5;
        int cbase = (nc0 & 31) + (lane & 15);
        v8f acc;
#pragma unroll
        for (int r = 0; r < 8; ++r) acc[r] = 0.0f;

        for (int kb = 0; kb < 22; ++kb) {
            int m = lane & 15;
            // gather V fragment first (global), then P fragment (LDS b128)
            v16h bfr;
#pragma unroll
            for (int r = 0; r < 8; ++r) {
                int jw0 = kb * 32 + kb0 + 2 * r;
                int t0 = kstokH[jw0 * 64 + tn];
                int t1 = kstokH[(jw0 + 1) * 64 + tn];
                bfr[2 * r]     = vh[(size_t)t0 * 32 + cbase];
                bfr[2 * r + 1] = vh[(size_t)t1 * 32 + cbase];
            }
            const f32x4* s0 = (const f32x4*)&S[m][kb * 32 + kb0];
            const f32x4* s1 = (const f32x4*)&S[m][kb * 32 + kb0 + 8];
            f32x4 pa = s0[0], pb = s0[1], pc = s1[0], pd = s1[1];
            v16h af;
#pragma unroll
            for (int i = 0; i < 4; ++i) {
                af[i]      = (_Float16)pa[i];
                af[4 + i]  = (_Float16)pb[i];
                af[8 + i]  = (_Float16)pc[i];
                af[12 + i] = (_Float16)pd[i];
            }
            acc = __builtin_amdgcn_wmma_f32_16x16x32_f16(
                false, af, false, bfr, (short)0, acc, false, false);
        }
        int n = lane & 15;
        int moff = (lane < 16) ? 0 : 8;
#pragma unroll
        for (int r = 0; r < 8; ++r) {
            int i = qt * 16 + r + moff;
            if (i < L) {
                int ch = nc0 + n;
                int tok = qtokH[i * ws2 + (ch >> 5)];
                attn16[((size_t)tok * NB + b) * DM + head * 32 + (ch & 31)] =
                    (_Float16)acc[r];
            }
        }
    }
}

// ---------------------------------------------------------------------------
// LayerNorm over 256 features, wave per row.
// ---------------------------------------------------------------------------
__global__ void k_layernorm(const float* __restrict__ y, const float* __restrict__ g,
                            const float* __restrict__ beta, float* __restrict__ outF,
                            _Float16* __restrict__ outH, int rows) {
    int wave = threadIdx.x >> 5, lane = threadIdx.x & 31;
    int row = blockIdx.x * (blockDim.x >> 5) + wave;
    if (row >= rows) return;
    const float* yr = y + (size_t)row * DM;
    float vals[8];
    float mu = 0.0f;
#pragma unroll
    for (int u = 0; u < 8; ++u) { vals[u] = yr[lane + u * 32]; mu += vals[u]; }
#pragma unroll
    for (int o = 16; o > 0; o >>= 1) mu += __shfl_xor(mu, o, 32);
    mu *= (1.0f / 256.0f);
    float var = 0.0f;
#pragma unroll
    for (int u = 0; u < 8; ++u) { float d = vals[u] - mu; var += d * d; }
#pragma unroll
    for (int o = 16; o > 0; o >>= 1) var += __shfl_xor(var, o, 32);
    var *= (1.0f / 256.0f);
    float rs = rsqrtf(var + 1e-5f);
#pragma unroll
    for (int u = 0; u < 8; ++u) {
        int c = lane + u * 32;
        float o = (vals[u] - mu) * rs * g[c] + beta[c];
        if (outF) outF[(size_t)row * DM + c] = o;
        if (outH) outH[(size_t)row * DM + c] = (_Float16)o;
    }
}

// ---------------------------------------------------------------------------
extern "C" void kernel_launch(void* const* d_in, const int* in_sizes, int n_in,
                              void* d_out, int out_size, void* d_ws, size_t ws_size,
                              hipStream_t stream) {
    const float* x    = (const float*)d_in[0];
    const float* Wqkv = (const float*)d_in[2];
    const float* bqkv = (const float*)d_in[3];
    const float* Wout = (const float*)d_in[4];
    const float* bout = (const float*)d_in[5];
    const float* W1   = (const float*)d_in[6];
    const float* b1   = (const float*)d_in[7];
    const float* W2   = (const float*)d_in[8];
    const float* b2   = (const float*)d_in[9];
    const float* ln1g = (const float*)d_in[10];
    const float* ln1b = (const float*)d_in[11];
    const float* ln2g = (const float*)d_in[12];
    const float* ln2b = (const float*)d_in[13];
    const float* relP[4] = {(const float*)d_in[14], (const float*)d_in[15],
                            (const float*)d_in[16], (const float*)d_in[17]};
    float* out = (float*)d_out;

    char* wp = (char*)d_ws;
    auto alloc = [&](size_t bytes) -> void* {
        void* p = (void*)wp;
        wp += (bytes + 255) & ~(size_t)255;
        return p;
    };
    _Float16* x16   = (_Float16*)alloc((size_t)NROW * DM * 2);
    _Float16* wqkvH = (_Float16*)alloc((size_t)DM * 3 * DM * 2);
    _Float16* woutH = (_Float16*)alloc((size_t)DM * DM * 2);
    _Float16* w1H   = (_Float16*)alloc((size_t)DM * DFF * 2);
    _Float16* w2H   = (_Float16*)alloc((size_t)DFF * DM * 2);
    _Float16* q16   = (_Float16*)alloc((size_t)NROW * DM * 2);
    _Float16* k16   = (_Float16*)alloc((size_t)NROW * DM * 2);
    _Float16* v16   = (_Float16*)alloc((size_t)NROW * DM * 2);
    _Float16* attn16= (_Float16*)alloc((size_t)NROW * DM * 2);
    float*    y1    = (float*)alloc((size_t)NROW * DM * 4);
    float*    x1    = (float*)alloc((size_t)NROW * DM * 4);
    _Float16* x1_16 = (_Float16*)alloc((size_t)NROW * DM * 2);
    _Float16* h16   = (_Float16*)alloc((size_t)NROW * DFF * 2);
    float*    y2    = (float*)alloc((size_t)NROW * DM * 4);
    int*      qtok  = (int*)alloc((size_t)8 * L_ALL * 4);
    int*      kstok = (int*)alloc((size_t)8 * L_ALL * 64 * 4);
    float*    srm   = (float*)alloc((size_t)8 * L_ALL * 4);

    k_build_tables<<<8, 256, 0, stream>>>(qtok, kstok, srm);
    k_cvt_f16<<<2048, 256, 0, stream>>>(x,    x16,   NROW * DM);
    k_cvt_f16<<<256,  256, 0, stream>>>(Wqkv, wqkvH, DM * 3 * DM);
    k_cvt_f16<<<128,  256, 0, stream>>>(Wout, woutH, DM * DM);
    k_cvt_f16<<<512,  256, 0, stream>>>(W1,   w1H,   DM * DFF);
    k_cvt_f16<<<512,  256, 0, stream>>>(W2,   w2H,   DFF * DM);

    // 1) QKV projection
    k_gemm_f16<<<dim3(3 * DM / 128, NROW / 64), 256, 0, stream>>>(
        x16, wqkvH, bqkv, NROW, 3 * DM, DM, 0,
        nullptr, nullptr, nullptr, q16, k16, v16);

    // 2) windowed attention per head
    const int wsArr[8] = {1, 2, 4, 8, 1, 2, 4, 8};
    for (int h = 0; h < 8; ++h) {
        int ws = wsArr[h];
        int L = L_ALL / (ws * ws);
        int qTiles = (L + 15) / 16;
        k_attention<<<dim3(NB * qTiles), 512, 0, stream>>>(
            q16, k16, v16, attn16, qtok, kstok, srm, relP[h & 3], h, ws, L);
    }

    // 3) out projection + residual; LN1
    k_gemm_f16<<<dim3(DM / 128, NROW / 64), 256, 0, stream>>>(
        attn16, woutH, bout, NROW, DM, DM, 1,
        y1, nullptr, x, nullptr, nullptr, nullptr);
    k_layernorm<<<NROW / 8, 256, 0, stream>>>(y1, ln1g, ln1b, x1, x1_16, NROW);

    // 4) FFN + LN2 -> d_out
    k_gemm_f16<<<dim3(DFF / 128, NROW / 64), 256, 0, stream>>>(
        x1_16, w1H, b1, NROW, DFF, DM, 2,
        nullptr, h16, nullptr, nullptr, nullptr, nullptr);
    k_gemm_f16<<<dim3(DM / 128, NROW / 64), 256, 0, stream>>>(
        h16, w2H, b2, NROW, DM, DFF, 3,
        y2, nullptr, x1, nullptr, nullptr, nullptr);
    k_layernorm<<<NROW / 8, 256, 0, stream>>>(y2, ln2g, ln2b, out, nullptr, NROW);
}
